// MultiheadSdpaDiff3_68788196213162
// MI455X (gfx1250) — matrix operand order repaired and checked by reading, not compiled, not run on hardware
//
#include <hip/hip_runtime.h>
#include <hip/hip_bf16.h>

// ---------------------------------------------------------------------------
// Differential multi-head attention forward, CDNA5 (gfx1250) WMMA pipeline.
// B=2, T=2048, E=2048, H=8 diff pairs, D=128, value width 2D=256.
// Uses v_wmma_f32_16x16x32_f16 for all contractions and
// global_load_async_to_lds_b128 (ASYNCcnt) for byte-identical tile staging.
// ---------------------------------------------------------------------------

#define T_LEN 2048
#define E_DIM 2048
#define B_SZ  2
#define H_PAIRS 8
#define D_HEAD 128
#define LAMBDA_INIT_F 0.7836057665f
#define ONE_MINUS_LI_F 0.2163942335f
#define SOFTMAX_SCALE 0.08838834764831845f  // 1/sqrt(128)

typedef _Float16 v16h __attribute__((ext_vector_type(16)));
typedef _Float16 v8h  __attribute__((ext_vector_type(8)));
typedef float    v8f  __attribute__((ext_vector_type(8)));

union Frag16 {
  v16h v;
  struct { v8h lo, hi; } p;
};

__device__ inline v8f vzero8() {
  v8f z = {0.f, 0.f, 0.f, 0.f, 0.f, 0.f, 0.f, 0.f};
  return z;
}

__device__ inline v8f wmma_f16(v16h a, v16h b, v8f c) {
  // D = A(16x32 f16) * B(32x16 f16) + C(16x16 f32)
  return __builtin_amdgcn_wmma_f32_16x16x32_f16(false, a, false, b,
                                                (short)0, c, false, false);
}

// Async global->LDS copy, 16 bytes per lane (CDNA5, tracked by ASYNCcnt).
// LDS destination is the per-lane vdst VGPR value; addr[31:0] of a flat shared pointer
// is the workgroup-relative LDS byte offset.
__device__ inline void async_g2l_b128(uint32_t lds_off, const _Float16* gptr) {
  asm volatile("global_load_async_to_lds_b128 %0, %1, off"
               :
               : "v"(lds_off), "v"(gptr)
               : "memory");
}

__device__ inline void wait_asynccnt0() {
  asm volatile("s_wait_asynccnt 0x0" ::: "memory");
}

__device__ inline uint32_t lds_off_of(const void* p) {
  return (uint32_t)(uintptr_t)p;
}

// ---------------------------------------------------------------------------
// fp32 -> fp16 conversion
// ---------------------------------------------------------------------------
__global__ void k_cvt_f16(const float* __restrict__ src,
                          _Float16* __restrict__ dst, int n) {
  int i = blockIdx.x * blockDim.x + threadIdx.x;
  if (i < n) dst[i] = (_Float16)src[i];
}

// ---------------------------------------------------------------------------
// lambda = exp(sum(lq1*lk1)) - exp(sum(lq2*lk2)) + LAMBDA_INIT
// ---------------------------------------------------------------------------
__global__ void k_lambda(const float* __restrict__ lq1, const float* __restrict__ lk1,
                         const float* __restrict__ lq2, const float* __restrict__ lk2,
                         float* __restrict__ lam) {
  __shared__ float s1[128];
  __shared__ float s2[128];
  int t = threadIdx.x;
  s1[t] = lq1[t] * lk1[t];
  s2[t] = lq2[t] * lk2[t];
  __syncthreads();
  for (int s = 64; s > 0; s >>= 1) {
    if (t < s) { s1[t] += s1[t + s]; s2[t] += s2[t + s]; }
    __syncthreads();
  }
  if (t == 0) *lam = __expf(s1[0]) - __expf(s2[0]) + LAMBDA_INIT_F;
}

// ---------------------------------------------------------------------------
// WMMA GEMM: C[M,N] = A[M,K] * B[K,N], A/B f16 row-major.
// Block tile 128x128, K-step 32, 256 threads = 8 waves (2x4), each wave 64x32.
// A tile staged via async-to-LDS; B tile transposed with paired b32 scatters.
// OUT_F16: store f16, else fp32.
// ---------------------------------------------------------------------------
template <int OUT_F16>
__global__ __launch_bounds__(256) void k_gemm(const _Float16* __restrict__ A,
                                              const _Float16* __restrict__ Bm,
                                              void* __restrict__ C,
                                              int M, int N, int K) {
  __shared__ _Float16 As[128 * 32];   // [m][k]
  __shared__ _Float16 Bs[128 * 32];   // transposed: [n][k]

  const int tid  = threadIdx.x;
  const int lane = tid & 31;
  const int wave = tid >> 5;
  const int wm   = (wave >> 2) * 64;  // wave M base within block
  const int wn   = (wave & 3) * 32;   // wave N base within block
  const int bm   = blockIdx.y * 128;
  const int bn   = blockIdx.x * 128;
  const int m16  = lane & 15;
  const int hi   = lane >> 4;
  const int kb   = hi * 8;            // A-frag K sub-offset
  const int ks   = hi * 16;           // unused; B-frag uses kb as well

  v8f acc[4][2];
#pragma unroll
  for (int mi = 0; mi < 4; ++mi)
#pragma unroll
    for (int ni = 0; ni < 2; ++ni) acc[mi][ni] = vzero8();

  for (int k0 = 0; k0 < K; k0 += 32) {
    __syncthreads();
    // Stage A tile 128x32 via async global->LDS (two b128 per thread)
#pragma unroll
    for (int i = 0; i < 2; ++i) {
      int idx = tid + i * 256;         // 0..511
      int row = idx >> 2, g4 = idx & 3;
      async_g2l_b128(lds_off_of(As + row * 32 + g4 * 8),
                     A + (size_t)(bm + row) * K + k0 + g4 * 8);
    }
    // Stage B tile 32x128 transposed into Bs[n][k]; two adjacent k-rows per
    // thread are packed to 32-bit words so scatters are ds_store_b32.
    {
      int kp = tid >> 4, g8 = tid & 15;   // kp: k-row pair 0..15
      v8h b0 = *(const v8h*)(Bm + (size_t)(k0 + kp * 2) * N + bn + g8 * 8);
      v8h b1 = *(const v8h*)(Bm + (size_t)(k0 + kp * 2 + 1) * N + bn + g8 * 8);
#pragma unroll
      for (int j = 0; j < 8; ++j) {
        union { _Float16 h[2]; uint32_t u; } pk;
        pk.h[0] = b0[j];
        pk.h[1] = b1[j];
        *(uint32_t*)(Bs + (g8 * 8 + j) * 32 + kp * 2) = pk.u;
      }
    }
    wait_asynccnt0();
    __syncthreads();

    Frag16 af[4], bf[2];
#pragma unroll
    for (int mi = 0; mi < 4; ++mi) {
      const _Float16* p = As + (wm + mi * 16 + m16) * 32;
      af[mi].p.lo = *(const v8h*)(p + kb);
      af[mi].p.hi = *(const v8h*)(p + 16 + kb);
    }
#pragma unroll
    for (int ni = 0; ni < 2; ++ni) {
      const _Float16* p = Bs + (wn + ni * 16 + m16) * 32 + kb;
      bf[ni].p.lo = *(const v8h*)(p);
      bf[ni].p.hi = *(const v8h*)(p + 16);
    }
#pragma unroll
    for (int mi = 0; mi < 4; ++mi)
#pragma unroll
      for (int ni = 0; ni < 2; ++ni)
        acc[mi][ni] = wmma_f16(af[mi].v, bf[ni].v, acc[mi][ni]);
  }

  // Epilogue: C/D layout — lane holds (m = r + 8*hi, n = m16) per 16x16 tile.
#pragma unroll
  for (int mi = 0; mi < 4; ++mi) {
#pragma unroll
    for (int ni = 0; ni < 2; ++ni) {
#pragma unroll
      for (int r = 0; r < 8; ++r) {
        int row = bm + wm + mi * 16 + r + 8 * hi;
        int col = bn + wn + ni * 16 + m16;
        if (OUT_F16)
          ((_Float16*)C)[(size_t)row * N + col] = (_Float16)acc[mi][ni][r];
        else
          ((float*)C)[(size_t)row * N + col] = acc[mi][ni][r];
      }
    }
  }
}

// ---------------------------------------------------------------------------
// Flash attention (causal), per block: one (b, h, c) and 128 query rows.
// 8 waves * 16 q-rows. KV tile = 64. Out: [b][t][h][256] fp32 (o1 or o2).
// K tile staged via async-to-LDS; V transposed with paired b32 scatters.
// ---------------------------------------------------------------------------
__global__ __launch_bounds__(256) void k_attn(const _Float16* __restrict__ Qh,
                                              const _Float16* __restrict__ Kh,
                                              const _Float16* __restrict__ Vh,
                                              float* __restrict__ Out1,
                                              float* __restrict__ Out2) {
  __shared__ _Float16 Ks[64 * 128];    // [key][d]
  __shared__ _Float16 Vs[256 * 64];    // transposed: [vcol][key]
  __shared__ _Float16 Ps[8 * 16 * 64]; // per-wave P scratch

  const int qt = blockIdx.x;          // query tile (0..15)
  const int y  = blockIdx.y;          // 0..31
  const int b  = y >> 4;
  const int h  = (y >> 1) & 7;
  const int c  = y & 1;
  float* __restrict__ Out = (c == 0) ? Out1 : Out2;

  const int tid  = threadIdx.x;
  const int lane = tid & 31;
  const int wave = tid >> 5;
  const int m16  = lane & 15;
  const int hi   = lane >> 4;
  const int kb   = hi * 8;
  const int ks   = hi * 16;

  const _Float16* Qb = Qh + (size_t)b * T_LEN * E_DIM + h * 256 + c * 128;
  const _Float16* Kb = Kh + (size_t)b * T_LEN * E_DIM + h * 256 + c * 128;
  const _Float16* Vb = Vh + (size_t)b * T_LEN * E_DIM + h * 256;

  const int qrowBase = qt * 128 + wave * 16;

  // Q fragments held in registers for the whole kernel: 16 rows x 128 (D).
  Frag16 aq[4];
  {
    const _Float16* qp = Qb + (size_t)(qrowBase + m16) * E_DIM;
#pragma unroll
    for (int kk = 0; kk < 4; ++kk) {
      aq[kk].p.lo = *(const v8h*)(qp + kk * 32 + kb);
      aq[kk].p.hi = *(const v8h*)(qp + kk * 32 + 16 + kb);
    }
  }

  float m_i[8], l_i[8];
#pragma unroll
  for (int r = 0; r < 8; ++r) { m_i[r] = -1e38f; l_i[r] = 0.f; }
  v8f oacc[16];
#pragma unroll
  for (int ni = 0; ni < 16; ++ni) oacc[ni] = vzero8();

  const int ktEnd = 2 * qt + 2;   // causal: keys up to qt*128+127
  for (int kt = 0; kt < ktEnd; ++kt) {
    const int kbase = kt * 64;
    __syncthreads();
    // Stage K tile 64x128 via async global->LDS (row-major, d contiguous)
#pragma unroll
    for (int i = 0; i < 4; ++i) {
      int idx = tid + i * 256;          // 0..1023
      int row = idx >> 4, g8 = idx & 15;
      async_g2l_b128(lds_off_of(Ks + row * 128 + g8 * 8),
                     Kb + (size_t)(kbase + row) * E_DIM + g8 * 8);
    }
    // Stage V tile 64x256 transposed into Vs[vcol][key]; two adjacent key
    // rows packed per thread so scatters are ds_store_b32.
#pragma unroll
    for (int i = 0; i < 4; ++i) {
      int idx = tid + i * 256;          // 0..1023
      int rp = idx >> 5, g8 = idx & 31; // rp: key-row pair 0..31
      v8h a0 = *(const v8h*)(Vb + (size_t)(kbase + rp * 2) * E_DIM + g8 * 8);
      v8h a1 = *(const v8h*)(Vb + (size_t)(kbase + rp * 2 + 1) * E_DIM + g8 * 8);
#pragma unroll
      for (int j = 0; j < 8; ++j) {
        union { _Float16 h[2]; uint32_t u; } pk;
        pk.h[0] = a0[j];
        pk.h[1] = a1[j];
        *(uint32_t*)(Vs + (g8 * 8 + j) * 64 + rp * 2) = pk.u;
      }
    }
    wait_asynccnt0();
    __syncthreads();

    // S = Q K^T (16 q-rows x 64 keys), 16 WMMAs
    v8f st[4];
#pragma unroll
    for (int nk = 0; nk < 4; ++nk) {
      v8f s = vzero8();
#pragma unroll
      for (int kk = 0; kk < 4; ++kk) {
        Frag16 bk;
        const _Float16* p = Ks + (nk * 16 + m16) * 128 + kk * 32 + kb;
        bk.p.lo = *(const v8h*)p;
        bk.p.hi = *(const v8h*)(p + 16);
        s = wmma_f16(aq[kk].v, bk.v, s);
      }
      st[nk] = s;
    }

    // Online softmax (row reductions across each 16-lane half)
    float corr[8];
#pragma unroll
    for (int r = 0; r < 8; ++r) {
      const int qrow = qrowBase + r + 8 * hi;
      float mx = -1e38f;
#pragma unroll
      for (int nk = 0; nk < 4; ++nk) {
        float xv = st[nk][r] * SOFTMAX_SCALE;
        if (kbase + nk * 16 + m16 > qrow) xv = -1e38f;
        st[nk][r] = xv;
        mx = fmaxf(mx, xv);
      }
      mx = fmaxf(mx, __shfl_xor(mx, 8, 16));
      mx = fmaxf(mx, __shfl_xor(mx, 4, 16));
      mx = fmaxf(mx, __shfl_xor(mx, 2, 16));
      mx = fmaxf(mx, __shfl_xor(mx, 1, 16));
      const float mn = fmaxf(m_i[r], mx);
      const float cr = __expf(m_i[r] - mn);
      float rs = 0.f;
#pragma unroll
      for (int nk = 0; nk < 4; ++nk) {
        float p = __expf(st[nk][r] - mn);
        st[nk][r] = p;
        rs += p;
      }
      rs += __shfl_xor(rs, 8, 16);
      rs += __shfl_xor(rs, 4, 16);
      rs += __shfl_xor(rs, 2, 16);
      rs += __shfl_xor(rs, 1, 16);
      l_i[r] = l_i[r] * cr + rs;
      m_i[r] = mn;
      corr[r] = cr;
    }
    // Rescale running output
#pragma unroll
    for (int ni = 0; ni < 16; ++ni)
#pragma unroll
      for (int r = 0; r < 8; ++r) oacc[ni][r] *= corr[r];

    // C-layout -> A-layout via per-wave LDS scratch
    _Float16* Pw = Ps + wave * (16 * 64);
#pragma unroll
    for (int nk = 0; nk < 4; ++nk)
#pragma unroll
      for (int r = 0; r < 8; ++r)
        Pw[(r + 8 * hi) * 64 + nk * 16 + m16] = (_Float16)st[nk][r];

    // O += P (16x64) * V (64x256): 32 WMMAs
#pragma unroll
    for (int kk2 = 0; kk2 < 2; ++kk2) {
      Frag16 ap;
      const _Float16* pp = Pw + m16 * 64 + kk2 * 32;
      ap.p.lo = *(const v8h*)(pp + kb);
      ap.p.hi = *(const v8h*)(pp + 16 + kb);
#pragma unroll
      for (int ni = 0; ni < 16; ++ni) {
        Frag16 bv;
        const _Float16* vp = Vs + (ni * 16 + m16) * 64 + kk2 * 32 + kb;
        bv.p.lo = *(const v8h*)vp;
        bv.p.hi = *(const v8h*)(vp + 16);
        oacc[ni] = wmma_f16(ap.v, bv.v, oacc[ni]);
      }
    }
  }

  // Finalize: divide by row sums, store fp32 [b][t][h][256]
#pragma unroll
  for (int ni = 0; ni < 16; ++ni) {
#pragma unroll
    for (int r = 0; r < 8; ++r) {
      int t = qrowBase + r + 8 * hi;
      int col = ni * 16 + m16;
      Out[(((size_t)(b * T_LEN + t)) * 8 + h) * 256 + col] = oacc[ni][r] / l_i[r];
    }
  }
}

// ---------------------------------------------------------------------------
// Combine: o = o1 - lam*o2; RMSNorm over 256; *g; *(1-LAMBDA_INIT); -> f16
// One wave per (b*T+t, head). Block = 8 waves = 8 heads, grid = B*T.
// ---------------------------------------------------------------------------
__global__ __launch_bounds__(256) void k_combine(const float* __restrict__ o1,
                                                 const float* __restrict__ o2,
                                                 const float* __restrict__ lamp,
                                                 const float* __restrict__ g,
                                                 _Float16* __restrict__ oh) {
  const int bt   = blockIdx.x;
  const int head = threadIdx.x >> 5;
  const int lane = threadIdx.x & 31;
  const float lam = *lamp;
  const float* p1 = o1 + ((size_t)bt * 8 + head) * 256;
  const float* p2 = o2 + ((size_t)bt * 8 + head) * 256;

  float vals[8];
  float ss = 0.f;
#pragma unroll
  for (int i = 0; i < 8; ++i) {
    int col = i * 32 + lane;
    float v = p1[col] - lam * p2[col];
    vals[i] = v;
    ss += v * v;
  }
  ss += __shfl_xor(ss, 16, 32);
  ss += __shfl_xor(ss, 8, 32);
  ss += __shfl_xor(ss, 4, 32);
  ss += __shfl_xor(ss, 2, 32);
  ss += __shfl_xor(ss, 1, 32);
  const float rn = rsqrtf(ss * (1.0f / 256.0f) + 1e-5f) * ONE_MINUS_LI_F;
#pragma unroll
  for (int i = 0; i < 8; ++i) {
    int col = i * 32 + lane;
    oh[(size_t)bt * E_DIM + head * 256 + col] =
        (_Float16)(vals[i] * rn * g[head * 256 + col]);
  }
}

// ---------------------------------------------------------------------------
// Host-side launch
// ---------------------------------------------------------------------------
extern "C" void kernel_launch(void* const* d_in, const int* in_sizes, int n_in,
                              void* d_out, int out_size, void* d_ws, size_t ws_size,
                              hipStream_t stream) {
  (void)in_sizes; (void)n_in; (void)out_size; (void)ws_size;
  const float* x   = (const float*)d_in[0];
  const float* Wq  = (const float*)d_in[1];
  const float* Wk  = (const float*)d_in[2];
  const float* Wv  = (const float*)d_in[3];
  const float* Wo  = (const float*)d_in[4];
  const float* lq1 = (const float*)d_in[5];
  const float* lk1 = (const float*)d_in[6];
  const float* lq2 = (const float*)d_in[7];
  const float* lk2 = (const float*)d_in[8];
  const float* g   = (const float*)d_in[9];
  float* out = (float*)d_out;

  const size_t MN = (size_t)B_SZ * T_LEN * E_DIM;     // 8,388,608
  const size_t WN = (size_t)E_DIM * E_DIM;            // 4,194,304

  // Workspace carve-out (256B aligned)
  char* base = (char*)d_ws;
  size_t off = 0;
  auto alloc = [&](size_t bytes) -> char* {
    char* p = base + off;
    off = (off + bytes + 255) & ~(size_t)255;
    return p;
  };
  _Float16* xh  = (_Float16*)alloc(MN * 2);
  _Float16* Wqh = (_Float16*)alloc(WN * 2);
  _Float16* Wkh = (_Float16*)alloc(WN * 2);
  _Float16* Wvh = (_Float16*)alloc(WN * 2);
  _Float16* Woh = (_Float16*)alloc(WN * 2);
  _Float16* Qh  = (_Float16*)alloc(MN * 2);
  _Float16* Kh  = (_Float16*)alloc(MN * 2);
  _Float16* Vh  = (_Float16*)alloc(MN * 2);
  float*    o1  = (float*)alloc(MN * 4);
  float*    o2  = (float*)alloc(MN * 4);
  _Float16* oh  = (_Float16*)alloc(MN * 2);
  float*    lam = (float*)alloc(256);

  // 1) fp32 -> fp16 conversions
  k_cvt_f16<<<(int)((MN + 255) / 256), 256, 0, stream>>>(x,  xh,  (int)MN);
  k_cvt_f16<<<(int)((WN + 255) / 256), 256, 0, stream>>>(Wq, Wqh, (int)WN);
  k_cvt_f16<<<(int)((WN + 255) / 256), 256, 0, stream>>>(Wk, Wkh, (int)WN);
  k_cvt_f16<<<(int)((WN + 255) / 256), 256, 0, stream>>>(Wv, Wvh, (int)WN);
  k_cvt_f16<<<(int)((WN + 255) / 256), 256, 0, stream>>>(Wo, Woh, (int)WN);

  // 2) Projections: Q/K/V = xh @ W (f16 out)
  const int M = B_SZ * T_LEN, N = E_DIM, K = E_DIM;
  dim3 ggrid(N / 128, M / 128);
  k_gemm<1><<<ggrid, 256, 0, stream>>>(xh, Wqh, (void*)Qh, M, N, K);
  k_gemm<1><<<ggrid, 256, 0, stream>>>(xh, Wkh, (void*)Kh, M, N, K);
  k_gemm<1><<<ggrid, 256, 0, stream>>>(xh, Wvh, (void*)Vh, M, N, K);

  // 3) lambda scalar
  k_lambda<<<1, 128, 0, stream>>>(lq1, lk1, lq2, lk2, lam);

  // 4) Causal flash attention, both differential components
  dim3 agrid(T_LEN / 128, B_SZ * H_PAIRS * 2);
  k_attn<<<agrid, 256, 0, stream>>>(Qh, Kh, Vh, o1, o2);

  // 5) Combine + RMSNorm -> f16 activations
  k_combine<<<B_SZ * T_LEN, 256, 0, stream>>>(o1, o2, lam, g, oh);

  // 6) Output projection (fp32 out)
  k_gemm<0><<<ggrid, 256, 0, stream>>>(oh, Woh, (void*)out, M, N, K);
}
